// RWKVAttention_76218489635288
// MI455X (gfx1250) — matrix operand order repaired and checked by reading, not compiled, not run on hardware
//
#include <hip/hip_runtime.h>
#include <hip/hip_bf16.h>
#include <math.h>

#define BDIM 2
#define TDIM 4096
#define CDIM 2048
#define HDIM 32
#define NDIM 64
#define BT (BDIM*TDIM)
#define LMIX 32
#define CH 16
#define NCHUNK (TDIM/CH)

typedef __attribute__((ext_vector_type(16))) __bf16 v16bf;
typedef __attribute__((ext_vector_type(8)))  float  v8f;
typedef __attribute__((ext_vector_type(4)))  float  f4v;

using bf16 = __hip_bfloat16;

__device__ inline float bf2f(bf16 x) { return __bfloat162float(x); }
__device__ inline bf16  f2bf(float x) { return __float2bfloat16(x); }

// ---------------- CDNA5 async / TDM helpers ----------------------------------
#if defined(__HIP_DEVICE_COMPILE__) && __has_builtin(__builtin_amdgcn_tensor_load_to_lds)
#define HAVE_TDM 1
#else
#define HAVE_TDM 0
#endif

__device__ inline void async_b128(unsigned int ldsoff, const void* gaddr) {
  // GLOBAL_LOAD_ASYNC_TO_LDS_B128 (tracked by ASYNCcnt)
  asm volatile("global_load_async_to_lds_b128 %0, %1, off"
               :: "v"(ldsoff), "v"(gaddr) : "memory");
}

template<int N>
__device__ inline void wait_async() {
  asm volatile("s_wait_asynccnt %0" :: "n"(N) : "memory");
}

#if HAVE_TDM
typedef __attribute__((ext_vector_type(4))) unsigned int v4u;
typedef __attribute__((ext_vector_type(8))) int v8i;
typedef __attribute__((ext_vector_type(4))) int v4i;

template<int N>
__device__ inline void wait_tensor() {
#if __has_builtin(__builtin_amdgcn_s_wait_tensorcnt)
  __builtin_amdgcn_s_wait_tensorcnt((short)N);
#else
  asm volatile("s_wait_tensorcnt %0" :: "n"(N) : "memory");
#endif
}

// 2-D TDM tile load: 64 f32 x 16 rows, row stride CDIM elements (D# per ISA 8.3/8.4)
__device__ inline void tdm_load_tile_f32(unsigned int lds_byte, const float* gaddr) {
  unsigned long long ga = (unsigned long long)(uintptr_t)gaddr;
  v4u g0;
  g0.x = 1u;                                   // count=1, user mode
  g0.y = lds_byte;                             // lds_addr
  g0.z = (unsigned int)ga;                     // global_addr[31:0]
  g0.w = (unsigned int)((ga >> 32) & 0x01FFFFFFull) | 0x80000000u; // addr[56:32] | type=2
  v8i g1;
  g1[0] = (2 << 16);                                   // data_size = 4B
  g1[1] = (int)(((unsigned)CDIM & 0xFFFFu) << 16);     // tensor_dim0 lo16
  g1[2] = (int)((((unsigned)CDIM >> 16) & 0xFFFFu) |
                (((unsigned)BT & 0xFFFFu) << 16));     // dim0 hi16 | dim1 lo16
  g1[3] = (int)((((unsigned)BT >> 16) & 0xFFFFu) |
                (64u << 16));                          // dim1 hi16 | tile_dim0=64
  g1[4] = CH;                                          // tile_dim1=16, tile_dim2=0
  g1[5] = CDIM;                                        // tensor_dim0_stride lo32
  g1[6] = 0;
  g1[7] = 0;
  v4i z4 = {0,0,0,0};
  v8i z8 = {0,0,0,0,0,0,0,0};
  __builtin_amdgcn_tensor_load_to_lds(g0, g1, z4, z4, z8, 0);  // 6-arg form
}
#endif

// ---------------- WMMA fragment helpers --------------------------------------
// A-fragment (16x32 bf16, row-major, interleaved K layout per ISA 7.12.2)
__device__ inline v16bf load_fragA(const bf16* base, int ld, int lane) {
  int m  = lane & 15;
  int kb = (lane >> 4) << 3;            // 0 or 8
  const bf16* p = base + (size_t)m * ld + kb;
  v16bf f;
  ((f4v*)&f)[0] = *(const f4v*)(p);      // K = kb..kb+7
  ((f4v*)&f)[1] = *(const f4v*)(p + 16); // K = kb+16..kb+23
  return f;
}

// B-fragment (32x16 bf16) loaded from B^T stored row-major ([N][K])
__device__ inline v16bf load_fragB(const bf16* baseT, int ld, int lane) {
  int n  = lane & 15;
  int ks = (lane >> 4) << 3;            // 0 or 8
  const bf16* p = baseT + (size_t)n * ld + ks;
  v16bf f;
  ((f4v*)&f)[0] = *(const f4v*)(p);
  ((f4v*)&f)[1] = *(const f4v*)(p + 16);
  return f;
}

__device__ inline v8f wmma_bf16(v16bf a, v16bf b, v8f c) {
  return __builtin_amdgcn_wmma_f32_16x16x32_bf16(false, a, false, b, (short)0, c, false, false);
}

// ---------------- weight prep: f32 [R][C] -> bf16 transposed [C][R] ----------
__global__ void k_transpose_w(const float* __restrict__ src, bf16* __restrict__ dst,
                              int R, int Cc) {
  int i = blockIdx.x * 256 + threadIdx.x;
  if (i >= R * Cc) return;
  int r = i / Cc, c = i % Cc;
  dst[(size_t)c * R + r] = f2bf(src[i]);
}

// ---------------- token shift + maa_x mix ------------------------------------
__global__ void k_prep_mixed(const float* __restrict__ hidden, const float* __restrict__ attn_x,
                             const float* __restrict__ maa_x, bf16* __restrict__ mixed) {
  size_t i = (size_t)blockIdx.x * 256 + threadIdx.x;   // over BT*C
  int    c = (int)(i & (CDIM - 1));
  size_t row = i >> 11;                                // /CDIM
  int t = (int)(row & (TDIM - 1));
  int b = (int)(row >> 12);
  float x  = hidden[i];
  float sh = (t == 0) ? attn_x[b * CDIM + c] : hidden[i - CDIM];
  mixed[i] = f2bf(x + (sh - x) * maa_x[c]);
}

// ---------------- generic GEMM (bf16 WMMA) + tanh ----------------------------
__global__ void k_gemm_tanh(const bf16* __restrict__ A, const bf16* __restrict__ Bt,
                            bf16* __restrict__ out, int K, int ldout) {
  int lane = threadIdx.x;
  size_t tm = (size_t)blockIdx.x * 16;
  int    tn = blockIdx.y * 16;
  v8f acc = {0.f,0.f,0.f,0.f,0.f,0.f,0.f,0.f};
  const bf16* arow = A  + tm * (size_t)K;
  const bf16* brow = Bt + (size_t)tn * K;
  for (int k = 0; k < K; k += 32) {
    v16bf a  = load_fragA(arow + k, K, lane);
    v16bf bb = load_fragB(brow + k, K, lane);
    acc = wmma_bf16(a, bb, acc);
  }
  int n = lane & 15, mb = (lane >> 4) << 3;
#pragma unroll
  for (int i = 0; i < 8; i++)
    out[(tm + mb + i) * (size_t)ldout + tn + n] = f2bf(tanhf(acc[i]));
}

// ---------------- LoRA-mix apply: m = xxx @ w2 (K=32, one WMMA per f) --------
__global__ void k_mix_combine(const float* __restrict__ hidden, const float* __restrict__ attn_x,
                              const bf16* __restrict__ xxx, const bf16* __restrict__ w2t,
                              const float* __restrict__ maa_w, const float* __restrict__ maa_k,
                              const float* __restrict__ maa_v, const float* __restrict__ maa_r,
                              bf16* __restrict__ xw, bf16* __restrict__ rb,
                              bf16* __restrict__ kb, bf16* __restrict__ vb) {
  int lane = threadIdx.x;
  size_t tm = (size_t)blockIdx.x * 16;
  int    tn = blockIdx.y * 16;
  v8f m[4];
#pragma unroll
  for (int f = 0; f < 4; f++) {           // mw, mk, mv, mr (mg is dead code)
    v16bf a  = load_fragA(xxx + tm * (5 * LMIX) + f * LMIX, 5 * LMIX, lane);
    v16bf bb = load_fragB(w2t + ((size_t)f * CDIM + tn) * LMIX, LMIX, lane);
    v8f z = {0.f,0.f,0.f,0.f,0.f,0.f,0.f,0.f};
    m[f] = wmma_bf16(a, bb, z);
  }
  int n = lane & 15, mb = (lane >> 4) << 3;
  int c = tn + n;
  float cw = maa_w[c], ck = maa_k[c], cv = maa_v[c], cr = maa_r[c];
#pragma unroll
  for (int i = 0; i < 8; i++) {
    size_t row = tm + mb + i;
    int t = (int)(row & (TDIM - 1));
    int b = (int)(row >> 12);
    size_t g = row * CDIM + c;
    float x  = hidden[g];
    float sh = (t == 0) ? attn_x[b * CDIM + c] : hidden[g - CDIM];
    float xx = sh - x;
    xw[g] = f2bf(x + xx * (cw + m[0][i]));
    kb[g] = f2bf(x + xx * (ck + m[1][i]));
    vb[g] = f2bf(x + xx * (cv + m[2][i]));
    rb[g] = f2bf(x + xx * (cr + m[3][i]));
  }
}

// ---------------- decay GEMM2 + log-decay ------------------------------------
__global__ void k_decay2(const bf16* __restrict__ dtmp, const bf16* __restrict__ dw2t,
                         const float* __restrict__ tdecay, float* __restrict__ ldout) {
  int lane = threadIdx.x;
  size_t tm = (size_t)blockIdx.x * 16;
  int    tn = blockIdx.y * 16;
  v8f acc = {0.f,0.f,0.f,0.f,0.f,0.f,0.f,0.f};
  for (int k = 0; k < 64; k += 32) {
    v16bf a  = load_fragA(dtmp + tm * 64 + k, 64, lane);
    v16bf bb = load_fragB(dw2t + (size_t)tn * 64 + k, 64, lane);
    acc = wmma_bf16(a, bb, acc);
  }
  int n = lane & 15, mb = (lane >> 4) << 3;
  float td = tdecay[tn + n];
#pragma unroll
  for (int i = 0; i < 8; i++)
    ldout[(tm + mb + i) * (size_t)CDIM + tn + n] = -__expf(td + acc[i]); // log(decay)
}

// ---------------- chunked WKV recurrence (WMMA + async double buffering) -----
// One block per (b,h). 4 waves, wave w owns value-columns [16w,16w+16).
__global__ __launch_bounds__(128) void k_wkv(const bf16* __restrict__ rB,
                                             const bf16* __restrict__ kB,
                                             const bf16* __restrict__ vB,
                                             const float* __restrict__ ldB,
                                             const float* __restrict__ s0,
                                             const float* __restrict__ faaaa,
                                             float* __restrict__ y) {
  __shared__ __attribute__((aligned(16))) bf16  rRaw[2][CH][NDIM];   // async staged
  __shared__ __attribute__((aligned(16))) bf16  kRaw[2][CH][NDIM];
  __shared__ __attribute__((aligned(16))) bf16  vRaw[2][CH][NDIM];
  __shared__ __attribute__((aligned(16))) float ldRaw[2][CH][NDIM];  // TDM staged
  __shared__ __attribute__((aligned(16))) bf16  sRt[CH][NDIM];       // r * exp(cs_{l-1})
  __shared__ __attribute__((aligned(16))) bf16  sKt[CH][NDIM];       // k * exp(-cs_l)
  __shared__ __attribute__((aligned(16))) bf16  sKhT[NDIM][32];      // (k*exp(cs15-cs_l))^T, K pad 32
  __shared__ __attribute__((aligned(16))) bf16  sVt[NDIM][32];       // V^T, K pad 32
  __shared__ __attribute__((aligned(16))) bf16  sPm[CH][32];         // masked P + diag, K pad 32
  __shared__ __attribute__((aligned(16))) bf16  sST[4][CH][NDIM];    // state^T per wave
  __shared__ float sCs15[NDIM];
  __shared__ float sDiag[CH];

  int tid = threadIdx.x, lane = tid & 31, wv = tid >> 5;
  int bh = blockIdx.x, b = bh >> 5, h = bh & 31;
  float u = faaaa[h];
  int colbase = h * NDIM;
  size_t rowbase = (size_t)b * TDIM;
  int vg = wv * 16;
  int nn = lane & 15, mb8 = (lane >> 4) << 3;

  const bf16*  rT0 = rB  + rowbase * CDIM + colbase;
  const bf16*  kT0 = kB  + rowbase * CDIM + colbase;
  const bf16*  vT0 = vB  + rowbase * CDIM + colbase;
  const float* dT0 = ldB + rowbase * CDIM + colbase;

  // per-thread async slice: 16B of one row
  int sl = tid >> 3;              // row 0..15
  int so = (tid & 7) * 8;         // 8 bf16 per 16B slice

  // zero the K-padding regions once
  for (int i = tid; i < NDIM * 16; i += 128) {
    int r0 = i >> 4, c0 = (i & 15) + 16;
    sKhT[r0][c0] = f2bf(0.f);
    sVt[r0][c0]  = f2bf(0.f);
  }
  for (int i = tid; i < CH * 16; i += 128)
    sPm[i >> 4][(i & 15) + 16] = f2bf(0.f);
  if (tid < CH) sDiag[tid] = 0.f;

  // initial state (f32 WMMA-C layout) + transposed bf16 copy in LDS
  v8f S[4];
  const float* s0p = s0 + (size_t)bh * NDIM * NDIM;  // [keydim][valdim]
#pragma unroll
  for (int i = 0; i < 4; i++)
#pragma unroll
    for (int j = 0; j < 8; j++) {
      int keyr = i * 16 + mb8 + j;
      float v0 = s0p[keyr * NDIM + vg + nn];
      S[i][j] = v0;
      sST[wv][nn][keyr] = f2bf(v0);
    }

  // prologue: stage chunk 0 asynchronously
  {
    size_t go = (size_t)sl * CDIM + so;
    async_b128((unsigned int)(uintptr_t)&rRaw[0][sl][so], rT0 + go);
    async_b128((unsigned int)(uintptr_t)&kRaw[0][sl][so], kT0 + go);
    async_b128((unsigned int)(uintptr_t)&vRaw[0][sl][so], vT0 + go);
#if HAVE_TDM
    if (wv == 0) tdm_load_tile_f32((unsigned int)(uintptr_t)&ldRaw[0][0][0], dT0);
#else
#pragma unroll
    for (int q = 0; q < 2; q++) {
      int s2 = tid + q * 128, l2 = s2 >> 4, o2 = (s2 & 15) * 4;
      async_b128((unsigned int)(uintptr_t)&ldRaw[0][l2][o2],
                 dT0 + (size_t)l2 * CDIM + o2);
    }
#endif
  }

  for (int ch = 0; ch < NCHUNK; ++ch) {
    int nb = ch & 1;
    // prefetch next chunk into the other buffer
    if (ch + 1 < NCHUNK) {
      int pb = nb ^ 1;
      size_t go = (size_t)(((ch + 1) * CH) + sl) * CDIM + so;
      async_b128((unsigned int)(uintptr_t)&rRaw[pb][sl][so], rT0 + go);
      async_b128((unsigned int)(uintptr_t)&kRaw[pb][sl][so], kT0 + go);
      async_b128((unsigned int)(uintptr_t)&vRaw[pb][sl][so], vT0 + go);
#if HAVE_TDM
      if (wv == 0)
        tdm_load_tile_f32((unsigned int)(uintptr_t)&ldRaw[pb][0][0],
                          dT0 + (size_t)(ch + 1) * CH * CDIM);
      wait_async<3>();                 // previous batch (chunk ch) complete
      if (wv == 0) wait_tensor<1>();
#else
#pragma unroll
      for (int q = 0; q < 2; q++) {
        int s2 = tid + q * 128, l2 = s2 >> 4, o2 = (s2 & 15) * 4;
        async_b128((unsigned int)(uintptr_t)&ldRaw[pb][l2][o2],
                   dT0 + (size_t)((ch + 1) * CH + l2) * CDIM + o2);
      }
      wait_async<5>();
#endif
    } else {
      wait_async<0>();
#if HAVE_TDM
      if (wv == 0) wait_tensor<0>();
#endif
    }
    __syncthreads();   // staged data visible; previous phase-4 LDS reads done

    // phase 2: per key-channel cumulative log-decay + scaling (threads 0..63)
    if (tid < NDIM) {
      int n = tid;
      float cs = 0.f;
      float ktil[CH];
#pragma unroll
      for (int l = 0; l < CH; l++) {
        float ldv  = ldRaw[nb][l][n];
        float rraw = bf2f(rRaw[nb][l][n]);
        float kraw = bf2f(kRaw[nb][l][n]);
        sVt[n][l]  = vRaw[nb][l][n];              // transpose V in LDS
        atomicAdd(&sDiag[l], rraw * kraw);        // plain r.k for the u-diagonal
        sRt[l][n] = f2bf(rraw * __expf(cs));      // exp(cs_{l-1})
        cs += ldv;
        float kt = kraw * __expf(-cs);
        ktil[l] = kt;
        sKt[l][n] = f2bf(kt);
      }
      sCs15[n] = cs;
      float e15 = __expf(cs);
#pragma unroll
      for (int l = 0; l < CH; l++) sKhT[n][l] = f2bf(ktil[l] * e15);
    }
    __syncthreads();

    // phase 3: wave 0 builds masked P = tril_strict(R~ K~^T) + diag(u * r.k)
    if (wv == 0) {
      v8f P = {0.f,0.f,0.f,0.f,0.f,0.f,0.f,0.f};
      for (int k0 = 0; k0 < NDIM; k0 += 32) {
        v16bf a  = load_fragA(&sRt[0][k0], NDIM, lane);
        v16bf bb = load_fragB(&sKt[0][k0], NDIM, lane);
        P = wmma_bf16(a, bb, P);
      }
#pragma unroll
      for (int j = 0; j < 8; j++) {
        int l = mb8 + j, s = nn;
        float val = (s < l) ? P[j] : ((s == l) ? u * sDiag[l] : 0.f);
        sPm[l][s] = f2bf(val);
      }
      if (lane < CH) sDiag[lane] = 0.f;           // re-zero for next chunk
    }
    __syncthreads();

    // phase 4: Y = R~ S0 + Pm V ; S = diag(exp(cs15)) S + K^T V
    v8f Y = {0.f,0.f,0.f,0.f,0.f,0.f,0.f,0.f};
    for (int k0 = 0; k0 < NDIM; k0 += 32) {
      v16bf a  = load_fragA(&sRt[0][k0], NDIM, lane);
      v16bf bb = load_fragB(&sST[wv][0][k0], NDIM, lane);
      Y = wmma_bf16(a, bb, Y);
    }
    v16bf bV = load_fragB(&sVt[vg][0], 32, lane);
    {
      v16bf a = load_fragA(&sPm[0][0], 32, lane);
      Y = wmma_bf16(a, bV, Y);
    }
#pragma unroll
    for (int j = 0; j < 8; j++) {
      int l = mb8 + j;
      y[(rowbase + ch * CH + l) * CDIM + colbase + vg + nn] = Y[j];
    }
#pragma unroll
    for (int i = 0; i < 4; i++) {
#pragma unroll
      for (int j = 0; j < 8; j++)
        S[i][j] *= __expf(sCs15[i * 16 + mb8 + j]);
      v16bf aK = load_fragA(&sKhT[i * 16][0], 32, lane);
      S[i] = wmma_bf16(aK, bV, S[i]);
    }
#pragma unroll
    for (int i = 0; i < 4; i++)                    // own-wave slice: no barrier needed
#pragma unroll
      for (int j = 0; j < 8; j++)
        sST[wv][nn][i * 16 + mb8 + j] = f2bf(S[i][j]);
    // loop-top __syncthreads() orders these writes/reads vs. next chunk
  }
}

extern "C" void kernel_launch(void* const* d_in, const int* in_sizes, int n_in,
                              void* d_out, int out_size, void* d_ws, size_t ws_size,
                              hipStream_t stream) {
  (void)in_sizes; (void)n_in; (void)out_size; (void)ws_size;
  const float* hidden  = (const float*)d_in[0];
  const float* attn_x  = (const float*)d_in[1];
  const float* attn_kv = (const float*)d_in[2];
  const float* maa_x   = (const float*)d_in[4];
  const float* maa_w   = (const float*)d_in[5];
  const float* maa_k   = (const float*)d_in[6];
  const float* maa_v   = (const float*)d_in[7];
  const float* maa_r   = (const float*)d_in[8];
  const float* w1      = (const float*)d_in[10];
  const float* w2      = (const float*)d_in[11];
  const float* tdecay  = (const float*)d_in[12];
  const float* dw1     = (const float*)d_in[13];
  const float* dw2     = (const float*)d_in[14];
  const float* faaaa   = (const float*)d_in[15];

  char* ws = (char*)d_ws;
  size_t off = 0;
  auto alloc = [&](size_t bytes) -> char* {
    char* p = ws + off; off += (bytes + 255) & ~(size_t)255; return p;
  };
  bf16*  mixed = (bf16*)alloc((size_t)BT * CDIM * 2);   // later reused as xw
  bf16*  xxx   = (bf16*)alloc((size_t)BT * 160 * 2);
  bf16*  rb    = (bf16*)alloc((size_t)BT * CDIM * 2);
  bf16*  kb    = (bf16*)alloc((size_t)BT * CDIM * 2);
  bf16*  vb    = (bf16*)alloc((size_t)BT * CDIM * 2);
  float* ldb   = (float*)alloc((size_t)BT * CDIM * 4);
  bf16*  dtmp  = (bf16*)alloc((size_t)BT * 64 * 2);
  bf16*  w1t   = (bf16*)alloc((size_t)160 * CDIM * 2);
  bf16*  w2t   = (bf16*)alloc((size_t)4 * CDIM * 32 * 2);
  bf16*  dw1t  = (bf16*)alloc((size_t)64 * CDIM * 2);
  bf16*  dw2t  = (bf16*)alloc((size_t)CDIM * 64 * 2);

  // weight transposes (tiny, once per launch)
  k_transpose_w<<<(CDIM * 160 + 255) / 256, 256, 0, stream>>>(w1, w1t, CDIM, 160);
  for (int f = 0; f < 4; ++f)
    k_transpose_w<<<(32 * CDIM + 255) / 256, 256, 0, stream>>>(
        w2 + (size_t)f * 32 * CDIM, w2t + (size_t)f * CDIM * 32, 32, CDIM);
  k_transpose_w<<<(CDIM * 64 + 255) / 256, 256, 0, stream>>>(dw1, dw1t, CDIM, 64);
  k_transpose_w<<<(64 * CDIM + 255) / 256, 256, 0, stream>>>(dw2, dw2t, 64, CDIM);

  k_prep_mixed<<<(BT * CDIM) / 256, 256, 0, stream>>>(hidden, attn_x, maa_x, mixed);
  {
    dim3 g(BT / 16, 160 / 16);
    k_gemm_tanh<<<g, 32, 0, stream>>>(mixed, w1t, xxx, CDIM, 160);
  }
  {
    dim3 g(BT / 16, CDIM / 16);
    k_mix_combine<<<g, 32, 0, stream>>>(hidden, attn_x, xxx, w2t,
                                        maa_w, maa_k, maa_v, maa_r,
                                        mixed /*xw overwrites mixed*/, rb, kb, vb);
  }
  {
    dim3 g(BT / 16, 64 / 16);
    k_gemm_tanh<<<g, 32, 0, stream>>>(mixed, dw1t, dtmp, CDIM, 64);
  }
  {
    dim3 g(BT / 16, CDIM / 16);
    k_decay2<<<g, 32, 0, stream>>>(dtmp, dw2t, tdecay, ldb);
  }
  k_wkv<<<BDIM * HDIM, 128, 0, stream>>>(rb, kb, vb, ldb, attn_kv, faaaa, (float*)d_out);
}
